// Model_39676907885209
// MI455X (gfx1250) — matrix operand order repaired and checked by reading, not compiled, not run on hardware
//
#include <hip/hip_runtime.h>

typedef __attribute__((ext_vector_type(16))) _Float16 v16h;
typedef __attribute__((ext_vector_type(8)))  _Float16 v8h;
typedef __attribute__((ext_vector_type(8)))  float    v8f;

#define SLICE (512 * 512)
#define INV_TEMP (1.0f / 0.12f)
#define QPAD 520   // 16x512 Q tile row stride (halves): 1040B = 260 dwords -> conflict-free
#define PPAD 136   // 16x128 P strip row stride (halves): 272B  = 68 dwords -> conflict-free

__device__ __forceinline__ v16h cat8(v8h lo, v8h hi) {
    return __builtin_shufflevector(lo, hi, 0,1,2,3,4,5,6,7,8,9,10,11,12,13,14,15);
}

// ---------------------------------------------------------------------------
// Kernel 1: fused conv1x1(pad=1) for keys (x2) and values (x3), f32 -> f16.
// ---------------------------------------------------------------------------
__global__ void __launch_bounds__(256)
prep_kv(const float* __restrict__ x1,
        const float* __restrict__ Wk, const float* __restrict__ bk,
        const float* __restrict__ Wv, const float* __restrict__ bv,
        _Float16* __restrict__ K16, _Float16* __restrict__ V16) {
    int idx = blockIdx.x * 256 + threadIdx.x;     // [0, 16*512*512)
    int b   = idx >> 18;
    int rem = idx & (SLICE - 1);
    int h   = rem >> 9;
    int w   = rem & 511;

    float c0 = 0.f, c1 = 0.f, c2 = 0.f;
    if (h >= 1 && h <= 510 && w >= 1 && w <= 510) {
        const float* p = x1 + ((long)(b * 3) * 510 + (h - 1)) * 510 + (w - 1);
        c0 = p[0];
        c1 = p[510 * 510];
        c2 = p[2 * 510 * 510];
    }
#pragma unroll
    for (int o = 0; o < 8; ++o) {
        float kv = bk[o] + Wk[o*3+0]*c0 + Wk[o*3+1]*c1 + Wk[o*3+2]*c2;
        float vv = bv[o] + Wv[o*3+0]*c0 + Wv[o*3+1]*c1 + Wv[o*3+2]*c2;
        int off = (b * 8 + o) * SLICE + rem;
        K16[off] = (_Float16)kv;
        V16[off] = (_Float16)vv;
    }
}

// ---------------------------------------------------------------------------
// Kernel 2: per-slice transpose V16 -> V16T via LDS 32x32 tiles.
// ---------------------------------------------------------------------------
__global__ void __launch_bounds__(256)
transpose_v(const _Float16* __restrict__ V16, _Float16* __restrict__ V16T) {
    __shared__ _Float16 tile[32][34];
    int slice = blockIdx.x >> 8;
    int t  = blockIdx.x & 255;
    int tr = t >> 4, tc = t & 15;
    int tx = threadIdx.x & 31, ty = threadIdx.x >> 5;
    long base = (long)slice * SLICE;
#pragma unroll
    for (int i = 0; i < 4; ++i)
        tile[ty + i*8][tx] = V16[base + (long)(tr*32 + ty + i*8) * 512 + tc*32 + tx];
    __syncthreads();
#pragma unroll
    for (int i = 0; i < 4; ++i)
        V16T[base + (long)(tc*32 + ty + i*8) * 512 + tr*32 + tx] = tile[tx][ty + i*8];
}

// ---------------------------------------------------------------------------
// Kernel 3: flash attention per (slice, 16-query-row tile). 8 waves / block.
// Q tile staged in LDS (bank-conflict-free padded). Per 128-key step: each
// wave computes one 16x16 S block (16 WMMAs), online softmax, f16 P strip to
// LDS, then 16x64 slice of P*V (16 WMMAs) with rotated global B fragments.
// ---------------------------------------------------------------------------
__global__ void __launch_bounds__(256)
flash_attn(const _Float16* __restrict__ K16, const _Float16* __restrict__ V16,
           const _Float16* __restrict__ V16T, float* __restrict__ out) {
    __shared__ __align__(16) _Float16 Qlds[16][QPAD];   // ~16.6KB
    __shared__ __align__(16) _Float16 Plds[16][PPAD];   // ~4.3KB
    __shared__ __align__(16) float redmax[16][8];
    __shared__ __align__(16) float redsum[16][8];

    const int slice = blockIdx.x >> 5;
    const int ht    = blockIdx.x & 31;
    const int tid   = threadIdx.x;
    const int wave  = tid >> 5;
    const int lane  = tid & 31;
    const int n     = lane & 15;
    const int hi    = lane >> 4;

    const _Float16* Q  = K16  + (long)slice * SLICE + (long)(ht * 16) * 512;
    const _Float16* V  = V16  + (long)slice * SLICE;
    const _Float16* VT = V16T + (long)slice * SLICE;

    // ---- stage Q tile into LDS: 1024 chunks of 8 halves, fully coalesced ----
#pragma unroll
    for (int it = 0; it < 4; ++it) {
        int c   = tid + it * 256;          // chunk id
        int row = c >> 6;                  // (c*8) >> 9
        int col = (c * 8) & 511;
        *(v8h*)(&Qlds[row][col]) = *(const v8h*)(Q + c * 8);
    }
    __syncthreads();

    v8f acc[4] = {v8f{}, v8f{}, v8f{}, v8f{}};
    float mrow[8], lrow[8];
#pragma unroll
    for (int r = 0; r < 8; ++r) { mrow[r] = -1e30f; lrow[r] = 0.f; }

    for (int kstep = 0; kstep < 4; ++kstep) {
        const int k0 = kstep * 128 + wave * 16;
        const _Float16* Brow = V + (long)(k0 + n) * 512 + hi * 16;
        const _Float16* BT0  = VT + (long)(wave * 64 + n) * 512 + kstep * 128 + hi * 16;

        if (kstep < 3)
            __builtin_prefetch((const void*)(Brow + 128 * 512), 0, 1);

        // ---- S = Q * V^T (contract over 512 w); rotate global B one ahead ----
        v8f s = {};
        v16h bcur = *(const v16h*)(Brow);
#pragma unroll
        for (int wc = 0; wc < 16; ++wc) {
            v16h bnext = bcur;
            if (wc < 15) bnext = *(const v16h*)(Brow + (wc + 1) * 32);
            v8h alo = *(const v8h*)(&Qlds[n][wc * 32 + hi * 16]);
            v8h ahi = *(const v8h*)(&Qlds[n][wc * 32 + hi * 16 + 8]);
            s = __builtin_amdgcn_wmma_f32_16x16x32_f16(false, cat8(alo, ahi), false,
                                                       bcur, (short)0, s, false, false);
            bcur = bnext;
        }

        // hoist first PV B fragment: latency overlaps entire softmax phase
        v16h bpv = *(const v16h*)(BT0);

        // ---- row max of this wave's 16x16 block ----
        float sc[8], rmax[8];
#pragma unroll
        for (int r = 0; r < 8; ++r) {
            float x = s[r] * INV_TEMP;
            sc[r] = x;
            x = fmaxf(x, __shfl_xor(x, 1));
            x = fmaxf(x, __shfl_xor(x, 2));
            x = fmaxf(x, __shfl_xor(x, 4));
            x = fmaxf(x, __shfl_xor(x, 8));
            rmax[r] = x;
        }
        if (n == 0) {
#pragma unroll
            for (int r = 0; r < 8; ++r) redmax[r + hi * 8][wave] = rmax[r];
        }
        __syncthreads();

        // ---- combine maxes across waves (2x ds_read_b128 per row) ----
        float newm[8], scalef[8];
#pragma unroll
        for (int r = 0; r < 8; ++r) {
            const float4* p = (const float4*)(&redmax[r + hi * 8][0]);
            float4 a0 = p[0], a1 = p[1];
            float sm = fmaxf(fmaxf(fmaxf(a0.x, a0.y), fmaxf(a0.z, a0.w)),
                             fmaxf(fmaxf(a1.x, a1.y), fmaxf(a1.z, a1.w)));
            float nm = fmaxf(mrow[r], sm);
            newm[r]   = nm;
            scalef[r] = __expf(mrow[r] - nm);
            mrow[r]   = nm;
        }

        // ---- P = exp(S - m) -> f16 strip in LDS; partial row sums ----
        float rsum[8];
#pragma unroll
        for (int r = 0; r < 8; ++r) {
            float p = __expf(sc[r] - newm[r]);
            Plds[r + hi * 8][wave * 16 + n] = (_Float16)p;
            float x = p;
            x += __shfl_xor(x, 1);
            x += __shfl_xor(x, 2);
            x += __shfl_xor(x, 4);
            x += __shfl_xor(x, 8);
            rsum[r] = x;
        }
        if (n == 0) {
#pragma unroll
            for (int r = 0; r < 8; ++r) redsum[r + hi * 8][wave] = rsum[r];
        }
#pragma unroll
        for (int r = 0; r < 8; ++r) {
            acc[0][r] *= scalef[r]; acc[1][r] *= scalef[r];
            acc[2][r] *= scalef[r]; acc[3][r] *= scalef[r];
        }
        __syncthreads();

#pragma unroll
        for (int r = 0; r < 8; ++r) {
            const float4* p = (const float4*)(&redsum[r + hi * 8][0]);
            float4 a0 = p[0], a1 = p[1];
            float ss = (a0.x + a0.y) + (a0.z + a0.w) + (a1.x + a1.y) + (a1.z + a1.w);
            lrow[r] = lrow[r] * scalef[r] + ss;
        }

        // ---- O += P * V : rotated global B fragments from V^T ----
#pragma unroll
        for (int t = 0; t < 4; ++t) {
#pragma unroll
            for (int ks = 0; ks < 4; ++ks) {
                int f = t * 4 + ks;
                v16h bnext = bpv;
                if (f < 15) {
                    int fn = f + 1;
                    bnext = *(const v16h*)(BT0 + (fn >> 2) * (16 * 512) + (fn & 3) * 32);
                }
                v8h alo = *(const v8h*)(&Plds[n][ks * 32 + hi * 16]);
                v8h ahi = *(const v8h*)(&Plds[n][ks * 32 + hi * 16 + 8]);
                acc[t] = __builtin_amdgcn_wmma_f32_16x16x32_f16(false, cat8(alo, ahi), false,
                                                                bpv, (short)0, acc[t],
                                                                false, false);
                bpv = bnext;
            }
        }
    }

    // ---- normalize by l and store ----
    float inv[8];
#pragma unroll
    for (int r = 0; r < 8; ++r) inv[r] = 1.0f / lrow[r];
    float* outp = out + (long)slice * SLICE + (long)(ht * 16) * 512;
#pragma unroll
    for (int t = 0; t < 4; ++t) {
        int wbase = wave * 64 + t * 16;
#pragma unroll
        for (int r = 0; r < 8; ++r) {
            int row = r + hi * 8;
            outp[(long)row * 512 + wbase + n] = acc[t][r] * inv[r];
        }
    }
}

// ---------------------------------------------------------------------------
extern "C" void kernel_launch(void* const* d_in, const int* in_sizes, int n_in,
                              void* d_out, int out_size, void* d_ws, size_t ws_size,
                              hipStream_t stream) {
    (void)in_sizes; (void)n_in; (void)out_size; (void)ws_size;
    const float* x1 = (const float*)d_in[0];
    const float* Wk = (const float*)d_in[1];
    const float* bk = (const float*)d_in[2];
    const float* Wv = (const float*)d_in[3];
    const float* bv = (const float*)d_in[4];
    float* out = (float*)d_out;

    _Float16* K16  = (_Float16*)d_ws;
    _Float16* V16  = K16 + (size_t)128 * SLICE;
    _Float16* V16T = V16 + (size_t)128 * SLICE;

    prep_kv<<<(16 * SLICE) / 256, 256, 0, stream>>>(x1, Wk, bk, Wv, bv, K16, V16);
    transpose_v<<<128 * 256, 256, 0, stream>>>(V16, V16T);
    flash_attn<<<128 * 32, 256, 0, stream>>>(K16, V16, V16T, out);
}
